// GADBase_44375602103217
// MI455X (gfx1250) — matrix operand order repaired and checked by reading, not compiled, not run on hardware
//
#include <hip/hip_runtime.h>
#include <hip/hip_bf16.h>
#include <stdint.h>

// ---------------------------------------------------------------------------
// GADBase diffusion super-resolution for MI455X (gfx1250, wave32).
//
//  * 1 init kernel:   cv/ch coefficients (written straight into d_out) and
//                     seed iterate (written into the y_pred slot of d_out).
//  * 128 step kernels: fused diffuse + block-mean adjust, ping-ponging
//                     between d_out's y_pred slot and a 2MB d_ws buffer so
//                     step 127 lands exactly in y_pred.
//
//  CDNA5 features used:
//   - TDM tensor_load_to_lds (D# built per ISA ch.8) + s_wait_tensorcnt
//     (6-arg builtin form: g0, g1, g2, g3, g4, cpol)
//   - v_wmma_f32_16x16x32_f16 for the 4x4 mean pooling (Z = R * X * R)
//   - global_prefetch via __builtin_prefetch
// ---------------------------------------------------------------------------

typedef __attribute__((ext_vector_type(16))) _Float16     v16h;
typedef __attribute__((ext_vector_type(8)))  float        v8f;
typedef __attribute__((ext_vector_type(4)))  unsigned int v4u;
typedef __attribute__((ext_vector_type(8)))  int          v8i;
typedef __attribute__((ext_vector_type(4)))  int          v4i;

#define IMG_H   512
#define IMG_W   512
#define SRC_H   128
#define SRC_W   128
#define NB      2          // batch
#define NSTEPS  128
#define TILE_W  128
#define TILE_H  16
#define PX      132        // sX pitch (132 % 64 == 4 -> conflict-free column gathers)
#define PY      17         // sY pitch (17 coprime with 64 banks)
#define LCONST  0.24f
#define KINV2   (1.0f / (0.03f * 0.03f))
#define EPSV    1e-8f

// ---------------------------------------------------------------------------
// Init: cv = g(mean_c |dI_v|), ch = g(mean_c |dI_h|) from [img, guide] (4 ch),
// and copy y_bicubic into the iterate buffer (= d_out y_pred slot).
// ---------------------------------------------------------------------------
__global__ __launch_bounds__(256) void gad_init(
    const float* __restrict__ guide, const float* __restrict__ ybic,
    float* __restrict__ img0, float* __restrict__ cv, float* __restrict__ ch)
{
  const int HW  = IMG_H * IMG_W;
  int idx = blockIdx.x * 256 + threadIdx.x;
  if (idx >= NB * HW) return;
  int b = idx / HW, p = idx - b * HW;
  int y = p / IMG_W, x = p - y * IMG_W;

  const float* yb = ybic  + (size_t)b * HW;
  const float* gd = guide + (size_t)b * 3 * HW;
  float c0 = yb[p];
  img0[idx] = c0;                               // seed iterate

  if (y < IMG_H - 1) {                          // vertical coefficient
    float d = fabsf(yb[p + IMG_W] - c0);
    #pragma unroll
    for (int c = 0; c < 3; ++c)
      d += fabsf(gd[c * HW + p + IMG_W] - gd[c * HW + p]);
    d *= 0.25f;
    cv[((size_t)b * (IMG_H - 1) + y) * IMG_W + x] =
        1.0f / (1.0f + d * d * KINV2);
  }
  if (x < IMG_W - 1) {                          // horizontal coefficient
    float d = fabsf(yb[p + 1] - c0);
    #pragma unroll
    for (int c = 0; c < 3; ++c)
      d += fabsf(gd[c * HW + p + 1] - gd[c * HW + p]);
    d *= 0.25f;
    ch[((size_t)b * IMG_H + y) * (IMG_W - 1) + x] =
        1.0f / (1.0f + d * d * KINV2);
  }
}

// ---------------------------------------------------------------------------
// One fused diffusion + adjust step. Grid (4, 32, 2), block 256 (8 waves).
// Each block owns a 128x16 tile; each wave owns a 16x16 subtile for pooling.
// ---------------------------------------------------------------------------
__global__ __launch_bounds__(256) void gad_step(
    const float* __restrict__ Iin,  float* __restrict__ Iout,
    const float* __restrict__ cv,   const float* __restrict__ ch,
    const float* __restrict__ src,  const float* __restrict__ msk)
{
  __shared__ float sI[18 * 130];            // tile + halo (TDM destination)
  __shared__ float sX[TILE_H * PX];         // diffused values
  __shared__ float sY[8 * 16 * PY];         // per-wave pooling scratch

  const int b   = blockIdx.z;
  const int tx0 = blockIdx.x * TILE_W;
  const int ty0 = blockIdx.y * TILE_H;
  // tile+halo clamped to image so the TDM never reads OOB
  const int lx0 = (tx0 > 0) ? tx0 - 1 : 0;
  const int ly0 = (ty0 > 0) ? ty0 - 1 : 0;
  const int lx1 = (tx0 + TILE_W < IMG_W) ? tx0 + TILE_W + 1 : IMG_W;
  const int ly1 = (ty0 + TILE_H < IMG_H) ? ty0 + TILE_H + 1 : IMG_H;
  const int lw  = lx1 - lx0, lh = ly1 - ly0;

  const float* Ib = Iin + (size_t)b * IMG_H * IMG_W;

  // warm the L2 path for this tile's source blocks (global_prefetch_b8)
  __builtin_prefetch(src + ((size_t)b * SRC_H + (ty0 >> 2)) * SRC_W + (tx0 >> 2), 0, 0);

#if __has_builtin(__builtin_amdgcn_tensor_load_to_lds)
  if (threadIdx.x < 32) {
    // ---- Tensor DMA descriptor (D#), groups per CDNA5 ISA ch.8 ----
    unsigned long long ga = (unsigned long long)(uintptr_t)(Ib + (size_t)ly0 * IMG_W + lx0);
    unsigned int ldsOff = (unsigned int)(uintptr_t)(&sI[0]);   // low 32b of LDS flat addr = byte offset
    const unsigned int TW = (unsigned int)lw, TH = (unsigned int)lh;
    const unsigned int DIM0 = IMG_W, DIM1 = IMG_H, STR1 = IMG_H * IMG_W;
    unsigned int g0_0 = 1u;                                            // count=1, user mode
    unsigned int g0_1 = ldsOff;                                        // lds_addr
    unsigned int g0_2 = (unsigned int)ga;                              // global_addr[31:0]
    unsigned int g0_3 = ((unsigned int)(ga >> 32) & 0x01FFFFFFu) | (2u << 30); // addr[56:32] | type=2
    int g1_0 = (int)(2u << 16);                                        // data_size = 4B
    int g1_1 = (int)((DIM0 & 0xFFFFu) << 16);                          // tensor_dim0 lo16
    int g1_2 = (int)(((DIM0 >> 16) & 0xFFFFu) | ((DIM1 & 0xFFFFu) << 16)); // dim0 hi | dim1 lo
    int g1_3 = (int)(((DIM1 >> 16) & 0xFFFFu) | (TW << 16));           // dim1 hi | tile_dim0
    int g1_4 = (int)(TH & 0xFFFFu);                                    // tile_dim1 (tile_dim2=0)
    int g1_5 = (int)DIM0;                                              // tensor_dim0_stride lo32
    int g1_6 = (int)((STR1 & 0xFFFFu) << 16);                          // stride0 hi | stride1 lo
    int g1_7 = (int)(STR1 >> 16);                                      // stride1[47:16]
    #define RFL(x) __builtin_amdgcn_readfirstlane((int)(x))
    v4u g0 = { (unsigned)RFL(g0_0), (unsigned)RFL(g0_1),
               (unsigned)RFL(g0_2), (unsigned)RFL(g0_3) };
    v8i g1 = { RFL(g1_0), RFL(g1_1), RFL(g1_2), RFL(g1_3),
               RFL(g1_4), RFL(g1_5), RFL(g1_6), RFL(g1_7) };
    v4i gz  = { 0, 0, 0, 0 };
    v8i gz8 = { 0, 0, 0, 0, 0, 0, 0, 0 };
    // 6-arg form (clang-23 / therock-10.0 HIP headers)
    __builtin_amdgcn_tensor_load_to_lds(g0, g1, gz, gz, gz8, 0);       // tensor_load_to_lds
    __builtin_amdgcn_s_wait_tensorcnt(0);                              // s_wait_tensorcnt 0
    #undef RFL
  }
#else
  for (int t = threadIdx.x; t < lw * lh; t += 256)
    sI[t] = Ib[(size_t)(ly0 + t / lw) * IMG_W + (lx0 + t % lw)];
#endif
  __syncthreads();

  // ---- diffusion: I[y] += tv(y) - tv(y-1) + th(x) - th(x-1), fluxes 0 at edges
  const float* cvb = cv + (size_t)b * (IMG_H - 1) * IMG_W;
  const float* chb = ch + (size_t)b * IMG_H * (IMG_W - 1);
  #pragma unroll
  for (int i = 0; i < 8; ++i) {
    int p  = threadIdx.x + i * 256;
    int ly = p >> 7, lx = p & 127;
    int gy = ty0 + ly, gx = tx0 + lx;
    int si = (gy - ly0) * lw + (gx - lx0);
    float c   = sI[si];
    float acc = c;
    if (gy + 1 < IMG_H) acc += LCONST * cvb[(size_t)gy * IMG_W + gx]       * (sI[si + lw] - c);
    if (gy > 0)         acc -= LCONST * cvb[(size_t)(gy - 1) * IMG_W + gx] * (c - sI[si - lw]);
    if (gx + 1 < IMG_W) acc += LCONST * chb[(size_t)gy * (IMG_W - 1) + gx]     * (sI[si + 1] - c);
    if (gx > 0)         acc -= LCONST * chb[(size_t)gy * (IMG_W - 1) + gx - 1] * (c - sI[si - 1]);
    sX[ly * PX + lx] = acc;
  }
  __syncthreads();

  // ---- 4x4 mean pooling via WMMA:  Z = R * X * R,  R[i][j] = (i/4 == j/4)
  const int wv   = threadIdx.x >> 5;
  const int lane = threadIdx.x & 31;
  const int half = lane >> 4;
  const int ln16 = lane & 15;
  const float* xt = &sX[wv * 16];           // wave's 16x16 subtile (cols 16*wv..)
  float*       yt = &sY[wv * 16 * PY];

  // A1 = R (16x32 f16, ISA A-layout), B1 = X zero-padded to 32x16 (B-layout)
  v16h a1, b1;
  #pragma unroll
  for (int s = 0; s < 16; ++s) {
    int vg = s >> 1;
    int k  = (vg < 4 ? 0 : 16) + ((vg & 3) * 2 + (s & 1)) + (half ? 8 : 0);
    a1[s] = (_Float16)((k < 16 && (ln16 >> 2) == (k >> 2)) ? 1.0f : 0.0f);
    int kb = (s < 8 ? 0 : 16) + half * 8 + (s & 7);
    b1[s] = (kb < 16) ? (_Float16)xt[kb * PX + ln16] : (_Float16)0.0f;
  }
  v8f w = {};
  w = __builtin_amdgcn_wmma_f32_16x16x32_f16(false, a1, false, b1, (short)0, w, false, false);

  // spill W (C/D layout) to LDS so it can be re-gathered in A-layout
  #pragma unroll
  for (int r = 0; r < 8; ++r) yt[(r + half * 8) * PY + ln16] = w[r];

  // A2 = W zero-padded (A-layout), B2 = R (B-layout)
  v16h a2, b2;
  #pragma unroll
  for (int s = 0; s < 16; ++s) {
    int vg = s >> 1;
    int k  = (vg < 4 ? 0 : 16) + ((vg & 3) * 2 + (s & 1)) + (half ? 8 : 0);
    a2[s] = (k < 16) ? (_Float16)yt[ln16 * PY + k] : (_Float16)0.0f;
    int kb = (s < 8 ? 0 : 16) + half * 8 + (s & 7);
    b2[s] = (_Float16)((kb < 16 && (kb >> 2) == (ln16 >> 2)) ? 1.0f : 0.0f);
  }
  v8f z = {};
  z = __builtin_amdgcn_wmma_f32_16x16x32_f16(false, a2, false, b2, (short)0, z, false, false);

  // ---- masked ratio + scale + store (D layout: m = r + 8*half, n = ln16)
  const float* srcb = src + (size_t)b * SRC_H * SRC_W;
  const float* mskb = msk + (size_t)b * SRC_H * SRC_W;
  float*       Ob   = Iout + (size_t)b * IMG_H * IMG_W;
  const int gx = tx0 + wv * 16 + ln16;
  const int bx = gx >> 2;
  #pragma unroll
  for (int r = 0; r < 8; ++r) {
    int   m    = r + half * 8;
    int   gy   = ty0 + m;
    int   by   = gy >> 2;
    float mean = z[r] * 0.0625f;                          // /16
    float sv   = srcb[by * SRC_W + bx];
    bool  inv  = mskb[by * SRC_W + bx] > 0.5f;
    float rat  = inv ? 1.0f : sv / (mean + EPSV);
    Ob[(size_t)gy * IMG_W + gx] = xt[m * PX + ln16] * rat;
  }
}

// ---------------------------------------------------------------------------
extern "C" void kernel_launch(void* const* d_in, const int* in_sizes, int n_in,
                              void* d_out, int out_size, void* d_ws, size_t ws_size,
                              hipStream_t stream) {
  (void)in_sizes; (void)n_in; (void)out_size; (void)ws_size;
  const float* guide = (const float*)d_in[0];   // [2,3,512,512]
  const float* ybic  = (const float*)d_in[1];   // [2,1,512,512]
  const float* src   = (const float*)d_in[2];   // [2,1,128,128]
  const float* msk   = (const float*)d_in[3];   // [2,1,128,128]

  float* out   = (float*)d_out;
  float* ypred = out;                                   // 2*512*512
  float* cv    = out + (size_t)NB * IMG_H * IMG_W;      // 2*511*512
  float* ch    = cv  + (size_t)NB * (IMG_H - 1) * IMG_W;// 2*512*511
  float* bufA  = (float*)d_ws;                          // 2MB ping buffer

  const int total = NB * IMG_H * IMG_W;
  gad_init<<<(total + 255) / 256, 256, 0, stream>>>(guide, ybic, ypred, cv, ch);

  // ping-pong: start in ypred; 128 (even) steps end back in ypred.
  dim3 grid(IMG_W / TILE_W, IMG_H / TILE_H, NB), blk(256);
  float* bufs[2] = { ypred, bufA };
  for (int i = 0; i < NSTEPS; ++i) {
    gad_step<<<grid, blk, 0, stream>>>(bufs[i & 1], bufs[(i + 1) & 1],
                                       cv, ch, src, msk);
  }
}
